// AVWGCN_3616362463696
// MI455X (gfx1250) — matrix operand order repaired and checked by reading, not compiled, not run on hardware
//
#include <hip/hip_runtime.h>
#include <hip/hip_bf16.h>
#include <stdint.h>

// ---------------------------------------------------------------------------
// AVWGCN (AGCRN graph conv) for MI455X / gfx1250, bf16 WMMA with f32 accum.
// Shapes: B=64, N=2048, C=64, E=16, O=64, CHEB_K=3.
// ---------------------------------------------------------------------------

#define NN   2048   // nodes
#define BB   64     // batch
#define CC   64     // dim_in
#define EE   16     // embed dim
#define OO   64     // dim_out
#define KC   192    // CHEB_K * CC
#define NBC  4096   // BB * CC  (packed feature width)
#define WROW 12288  // KC * OO  (per-node weight row)

typedef __attribute__((ext_vector_type(16))) __bf16     v16bf;
typedef __attribute__((ext_vector_type(8)))  float      v8f;
typedef __attribute__((ext_vector_type(4)))  unsigned   v4u;

union Frag16 { v16bf v; v4u q[2]; };

static __device__ __forceinline__ unsigned short f32_to_bf16_rte(float f) {
    union { float f; unsigned u; } v; v.f = f;
    unsigned r = v.u + 0x7FFFu + ((v.u >> 16) & 1u);   // round-to-nearest-even
    return (unsigned short)(r >> 16);
}
static __device__ __forceinline__ float bf16_to_f32(unsigned short h) {
    union { unsigned u; float f; } v; v.u = ((unsigned)h) << 16;
    return v.f;
}
static __device__ __forceinline__ v8f wmma_bf16(const Frag16& a, const Frag16& b, v8f c) {
    // D = A(16x32 bf16) * B(32x16 bf16) + C(16x16 f32)
    return __builtin_amdgcn_wmma_f32_16x16x32_bf16(
        /*neg_a=*/false, a.v, /*neg_b=*/false, b.v,
        /*c_mod=*/(short)0, c, /*reuse_a=*/false, /*reuse_b=*/false);
}

// ---------------------------------------------------------------------------
// 1) supports = softmax(relu(E E^T)) row-wise  ->  bf16 [NN, NN]
//    one 256-thread block per row; 8 columns per thread.
// ---------------------------------------------------------------------------
__global__ __launch_bounds__(256)
void k_supports(const float* __restrict__ E, unsigned short* __restrict__ A_bf) {
    __shared__ float red[256];
    const int n   = blockIdx.x;
    const int tid = threadIdx.x;

    float en[EE];
#pragma unroll
    for (int d = 0; d < EE; ++d) en[d] = E[n * EE + d];

    float vals[8];
    float lmax = -3.0e38f;
#pragma unroll
    for (int i = 0; i < 8; ++i) {
        const int m = tid + i * 256;
        const float* em = E + m * EE;
        float s = 0.0f;
#pragma unroll
        for (int d = 0; d < EE; ++d) s += en[d] * em[d];
        s = fmaxf(s, 0.0f);               // relu
        vals[i] = s;
        lmax = fmaxf(lmax, s);
    }
    red[tid] = lmax; __syncthreads();
    for (int s = 128; s > 0; s >>= 1) {
        if (tid < s) red[tid] = fmaxf(red[tid], red[tid + s]);
        __syncthreads();
    }
    const float bmax = red[0];
    __syncthreads();

    float lsum = 0.0f;
#pragma unroll
    for (int i = 0; i < 8; ++i) {
        vals[i] = __expf(vals[i] - bmax);
        lsum += vals[i];
    }
    red[tid] = lsum; __syncthreads();
    for (int s = 128; s > 0; s >>= 1) {
        if (tid < s) red[tid] += red[tid + s];
        __syncthreads();
    }
    const float inv = 1.0f / red[0];

#pragma unroll
    for (int i = 0; i < 8; ++i) {
        const int m = tid + i * 256;
        A_bf[(size_t)n * NN + m] = f32_to_bf16_rte(vals[i] * inv);
    }
}

// ---------------------------------------------------------------------------
// 2) pack X:  Xb[m, b*64+c] = bf16(x[b, m, c])   ([NN, NBC] bf16)
// ---------------------------------------------------------------------------
__global__ __launch_bounds__(256)
void k_pack_x(const float* __restrict__ x, unsigned short* __restrict__ Xb) {
    const int idx = blockIdx.x * 256 + threadIdx.x;      // over NN*NBC = 8388608
    const int m = idx >> 12;
    const int r = idx & 4095;
    const int b = r >> 6;
    const int c = r & 63;
    Xb[idx] = f32_to_bf16_rte(x[((size_t)b * NN + m) * CC + c]);
}

// ---------------------------------------------------------------------------
// 3) GEMM:  D[M,NBC] = bf16( alpha * A[M,Kt] @ Bm[Kt,NBC] + beta * Cin )
//    M = Kt = 2048, NBC = 4096.  Block tile 128x128, 8 waves (2x4),
//    K-step 32 (one WMMA depth).  LDS staged, b128 moves.
// ---------------------------------------------------------------------------
__global__ __launch_bounds__(256)
void k_gemm_bf16(const unsigned short* __restrict__ A,
                 const unsigned short* __restrict__ Bm,
                 const unsigned short* __restrict__ Cin,
                 unsigned short* __restrict__ D,
                 float alpha, float beta) {
    const int Kt = 2048, Nn = NBC;
    __shared__ unsigned short sA[128 * 32];   // [row][k]  8 KB
    __shared__ unsigned short sB[32 * 128];   // [k][col]  8 KB

    const int tid  = threadIdx.x;
    const int wave = tid >> 5;
    const int lane = tid & 31;
    const int l15  = lane & 15;
    const int lh   = lane >> 4;
    const int wm   = wave >> 2;               // 0..1 : 64 rows each
    const int wn   = wave & 3;                // 0..3 : 32 cols each
    const int bM   = blockIdx.y * 128;
    const int bN   = blockIdx.x * 128;

    v8f acc[4][2];
#pragma unroll
    for (int mt = 0; mt < 4; ++mt)
#pragma unroll
        for (int nt = 0; nt < 2; ++nt) acc[mt][nt] = (v8f)0.0f;

    for (int k0 = 0; k0 < Kt; k0 += 32) {
        // ---- stage A tile 128x32: 4 threads/row x 16B, 2 passes
        {
            const int r  = tid >> 2;
            const int cq = tid & 3;
#pragma unroll
            for (int p = 0; p < 2; ++p) {
                const int row = r + p * 64;
                const v4u* src = (const v4u*)(A + (size_t)(bM + row) * Kt + k0) + cq;
                *((v4u*)(sA + row * 32) + cq) = *src;
            }
        }
        // ---- stage B tile 32x128: 16 threads/row x 16B, 2 passes
        {
            const int r  = tid >> 4;
            const int cq = tid & 15;
#pragma unroll
            for (int p = 0; p < 2; ++p) {
                const int row = r + p * 16;
                const v4u* src = (const v4u*)(Bm + (size_t)(k0 + row) * Nn + bN) + cq;
                *((v4u*)(sB + row * 128) + cq) = *src;
            }
        }
        if (k0 + 32 < Kt) {   // hint next tiles toward the caches (global_prefetch_b8)
            __builtin_prefetch(A + (size_t)(bM + (tid >> 1)) * Kt + (k0 + 32), 0, 1);
            __builtin_prefetch(Bm + (size_t)(k0 + 32 + (tid & 31)) * Nn + bN, 0, 1);
        }
        __syncthreads();

        // ---- B fragments (lane = N col; 16 K values per frag, table order)
        Frag16 fb[2];
#pragma unroll
        for (int nt = 0; nt < 2; ++nt) {
            const int col  = wn * 32 + nt * 16 + l15;
#pragma unroll
            for (int s = 0; s < 16; ++s) {
                const int krow = (s < 8) ? (8 * lh + s) : (16 + 8 * lh + (s - 8));
                fb[nt].v[s] = __builtin_bit_cast(__bf16, sB[krow * 128 + col]);
            }
        }
        // ---- A fragments (lane = M row; K split by lane-half) + WMMA
#pragma unroll
        for (int mt = 0; mt < 4; ++mt) {
            Frag16 fa;
            const int row = wm * 64 + mt * 16 + l15;
            fa.q[0] = *(const v4u*)(sA + row * 32 + 8 * lh);
            fa.q[1] = *(const v4u*)(sA + row * 32 + 16 + 8 * lh);
#pragma unroll
            for (int nt = 0; nt < 2; ++nt)
                acc[mt][nt] = wmma_bf16(fa, fb[nt], acc[mt][nt]);
        }
        __syncthreads();
    }

    // ---- epilogue: D = bf16(alpha*acc + beta*Cin)
#pragma unroll
    for (int mt = 0; mt < 4; ++mt) {
#pragma unroll
        for (int nt = 0; nt < 2; ++nt) {
            const int col = bN + wn * 32 + nt * 16 + l15;
#pragma unroll
            for (int v = 0; v < 8; ++v) {
                const int row = bM + wm * 64 + mt * 16 + v + 8 * lh;
                float r = alpha * acc[mt][nt][v];
                if (beta != 0.0f)
                    r += beta * bf16_to_f32(Cin[(size_t)row * Nn + col]);
                D[(size_t)row * Nn + col] = f32_to_bf16_rte(r);
            }
        }
    }
}

// ---------------------------------------------------------------------------
// 4) per-node weights:  Wn[n, (k*64+i)*64+o] = bf16( sum_d E[n,d]*Wpool[d,k,i,o] )
// ---------------------------------------------------------------------------
__global__ __launch_bounds__(256)
void k_weights(const float* __restrict__ E, const float* __restrict__ Wpool,
               unsigned short* __restrict__ Wn) {
    const int idx = blockIdx.x * 256 + threadIdx.x;      // over NN*WROW = 25165824
    const int n = idx / WROW;
    const int r = idx - n * WROW;
    float s = 0.0f;
#pragma unroll
    for (int d = 0; d < EE; ++d) s += E[n * EE + d] * Wpool[(size_t)d * WROW + r];
    Wn[idx] = f32_to_bf16_rte(s);
}

// 5) bias:  biasN[n,o] = sum_d E[n,d] * bias_pool[d,o]   (f32)
__global__ __launch_bounds__(256)
void k_bias(const float* __restrict__ E, const float* __restrict__ bp,
            float* __restrict__ biasN) {
    const int idx = blockIdx.x * 256 + threadIdx.x;      // over NN*OO = 131072
    const int n = idx >> 6;
    const int o = idx & 63;
    float s = 0.0f;
#pragma unroll
    for (int d = 0; d < EE; ++d) s += E[n * EE + d] * bp[d * OO + o];
    biasN[idx] = s;
}

// ---------------------------------------------------------------------------
// 6) per-node grouped GEMM: out[b,n,o] = sum_{k,c} xg[b,n,k,c]*W[n,k,c,o] + bias
//    one block (4 waves) per node; A=64x192 bf16, B=192x64 bf16 in LDS.
// ---------------------------------------------------------------------------
__global__ __launch_bounds__(128)
void k_node_gemm(const unsigned short* __restrict__ Xb,   // k=0 features
                 const unsigned short* __restrict__ Y1,   // k=1
                 const unsigned short* __restrict__ Y2,   // k=2
                 const unsigned short* __restrict__ Wn,
                 const float* __restrict__ biasN,
                 float* __restrict__ out) {
    __shared__ unsigned short sA[BB * KC];   // [b][k*64+c]  24 KB
    __shared__ unsigned short sB[KC * OO];   // [k*64+i][o]  24 KB

    const int n   = blockIdx.x;
    const int tid = threadIdx.x;

    // stage A: three [64 x 64] bf16 chunks, 8KB each -> 4 x b128 per thread
    const unsigned short* bufs[3] = { Xb + (size_t)n * NBC,
                                      Y1 + (size_t)n * NBC,
                                      Y2 + (size_t)n * NBC };
#pragma unroll
    for (int k = 0; k < 3; ++k) {
        const v4u* src = (const v4u*)bufs[k];
#pragma unroll
        for (int p = 0; p < 4; ++p) {
            const int idx = tid + p * 128;       // 0..511 v4u (8 bf16 each)
            const int b   = idx >> 3;
            const int c8  = idx & 7;
            *((v4u*)(sA + b * KC + k * 64) + c8) = src[idx];
        }
    }
    // stage B: 24KB contiguous -> 12 x b128 per thread
    {
        const v4u* src = (const v4u*)(Wn + (size_t)n * WROW);
        v4u* dst = (v4u*)sB;
#pragma unroll
        for (int p = 0; p < 12; ++p) dst[tid + p * 128] = src[tid + p * 128];
    }
    __syncthreads();

    const int wave = tid >> 5;       // M-tile (16 b-rows) per wave
    const int lane = tid & 31;
    const int l15  = lane & 15;
    const int lh   = lane >> 4;

    v8f acc[4];
#pragma unroll
    for (int nt = 0; nt < 4; ++nt) acc[nt] = (v8f)0.0f;

#pragma unroll
    for (int ks = 0; ks < 6; ++ks) {           // K = 192 = 6 * 32
        Frag16 fa;
        const int row = wave * 16 + l15;
        fa.q[0] = *(const v4u*)(sA + row * KC + ks * 32 + 8 * lh);
        fa.q[1] = *(const v4u*)(sA + row * KC + ks * 32 + 16 + 8 * lh);
#pragma unroll
        for (int nt = 0; nt < 4; ++nt) {
            Frag16 fb;
#pragma unroll
            for (int s = 0; s < 16; ++s) {
                const int krow = ks * 32 + ((s < 8) ? (8 * lh + s) : (16 + 8 * lh + (s - 8)));
                fb.v[s] = __builtin_bit_cast(__bf16, sB[krow * OO + nt * 16 + l15]);
            }
            acc[nt] = wmma_bf16(fa, fb, acc[nt]);
        }
    }

#pragma unroll
    for (int nt = 0; nt < 4; ++nt) {
        const int o = nt * 16 + l15;
        const float bias = biasN[n * OO + o];
#pragma unroll
        for (int v = 0; v < 8; ++v) {
            const int b = wave * 16 + v + 8 * lh;
            out[((size_t)b * NN + n) * OO + o] = acc[nt][v] + bias;
        }
    }
}

// ---------------------------------------------------------------------------
// launch
// ---------------------------------------------------------------------------
extern "C" void kernel_launch(void* const* d_in, const int* in_sizes, int n_in,
                              void* d_out, int out_size, void* d_ws, size_t ws_size,
                              hipStream_t stream) {
    (void)in_sizes; (void)n_in; (void)out_size; (void)ws_size;

    const float* x  = (const float*)d_in[0];   // [B,N,C]
    const float* E  = (const float*)d_in[1];   // [N,E]
    const float* Wp = (const float*)d_in[2];   // [E,K,C,O]
    const float* bp = (const float*)d_in[3];   // [E,O]
    float* out = (float*)d_out;                // [B,N,O]

    char* w = (char*)d_ws;
    unsigned short* A_bf  = (unsigned short*)(w);                       //  8 MB
    unsigned short* Xb    = (unsigned short*)(w + 8388608u);            // 16 MB
    unsigned short* Y1    = (unsigned short*)(w + 25165824u);           // 16 MB
    unsigned short* Y2    = (unsigned short*)(w + 41943040u);           // 16 MB
    unsigned short* Wn    = (unsigned short*)(w + 58720256u);           // 48 MB
    float*          biasN = (float*)         (w + 109051904u);          // 0.5 MB

    // independent prep
    k_supports<<<NN, 256, 0, stream>>>(E, A_bf);
    k_pack_x  <<<(NN * NBC) / 256, 256, 0, stream>>>(x, Xb);
    k_weights <<<(NN * WROW) / 256, 256, 0, stream>>>(E, Wp, Wn);
    k_bias    <<<(NN * OO) / 256, 256, 0, stream>>>(E, bp, biasN);

    // x_g1 = A @ X ;  x_g2 = 2*A @ x_g1 - X   (Chebyshev without forming A^2)
    dim3 ggrid(NBC / 128, NN / 128);
    k_gemm_bf16<<<ggrid, 256, 0, stream>>>(A_bf, Xb, nullptr, Y1, 1.0f,  0.0f);
    k_gemm_bf16<<<ggrid, 256, 0, stream>>>(A_bf, Y1, Xb,      Y2, 2.0f, -1.0f);

    // per-node grouped GEMM + bias
    k_node_gemm<<<NN, 128, 0, stream>>>(Xb, Y1, Y2, Wn, biasN, out);
}
